// KeyframeGenerator_60129542144368
// MI455X (gfx1250) — matrix operand order repaired and checked by reading, not compiled, not run on hardware
//
#include <hip/hip_runtime.h>
#include <hip/hip_bf16.h>
#include <math.h>

#define D_MOTION 135
#define D_MODEL  512
#define N_LAYERS 6
#define N_HEADS  8
#define D_HEAD   64
#define D_FF     2048
#define WIN_HALF 15
#define CTX      10
#define BB       8
#define TT       512
#define MTOK     (BB*TT)          // 4096 tokens

typedef __attribute__((ext_vector_type(16))) _Float16 v16h;
typedef __attribute__((ext_vector_type(8)))  _Float16 v8h;
typedef __attribute__((ext_vector_type(8)))  float    v8f;

enum { ACT_NONE=0, ACT_PRELU=1, ACT_RELU=2, ACT_SILU=3 };

// ---------------------------------------------------------------------------
// Prep: motion = x[...,:135]*mask + noise*(1-mask); traj = x[...,135:138]
// ---------------------------------------------------------------------------
__global__ void prep_kernel(const float* __restrict__ x, const float* __restrict__ noise,
                            float* __restrict__ motion_in, float* __restrict__ traj_in) {
    int idx = blockIdx.x * blockDim.x + threadIdx.x;
    const int total = MTOK * 138;
    if (idx >= total) return;
    int m = idx / 138, c = idx % 138;
    int t = m % TT;
    float xv = x[idx];
    if (c < D_MOTION) {
        bool keep = (t < CTX) || (t == TT - 1);
        float nv = noise[(size_t)m * D_MOTION + c];
        motion_in[(size_t)m * D_MOTION + c] = keep ? xv : nv;
    } else {
        traj_in[(size_t)m * 3 + (c - D_MOTION)] = xv;
    }
}

// ---------------------------------------------------------------------------
// LayerNorm: one wave (32 lanes) per row of 512
// ---------------------------------------------------------------------------
__global__ __launch_bounds__(256)
void ln_kernel(const float* __restrict__ X, const float* __restrict__ g,
               const float* __restrict__ b, float* __restrict__ Y) {
    int row  = blockIdx.x * 8 + (threadIdx.x >> 5);
    int lane = threadIdx.x & 31;
    const float* xr = X + (size_t)row * D_MODEL;
    float vals[16];
    float s = 0.f;
    #pragma unroll
    for (int i = 0; i < 16; i++) { vals[i] = xr[lane + 32*i]; s += vals[i]; }
    #pragma unroll
    for (int off = 16; off > 0; off >>= 1) s += __shfl_xor(s, off, 32);
    float mean = s * (1.f / D_MODEL);
    float v = 0.f;
    #pragma unroll
    for (int i = 0; i < 16; i++) { float d = vals[i] - mean; v += d * d; }
    #pragma unroll
    for (int off = 16; off > 0; off >>= 1) v += __shfl_xor(v, off, 32);
    float rstd = rsqrtf(v * (1.f / D_MODEL) + 1e-5f);
    float* yr = Y + (size_t)row * D_MODEL;
    #pragma unroll
    for (int i = 0; i < 16; i++) {
        int c = lane + 32*i;
        yr[c] = (vals[i] - mean) * rstd * g[c] + b[c];
    }
}

// ---------------------------------------------------------------------------
// WMMA GEMM: C[M,N] = act( A[M,K] @ W[N,K]^T + bias + res )
// Block: 256 thr = 8 waves; block tile 128(M) x 64(N).
// Wave w owns rows [w*16, w*16+16) and all 4 N-subtiles (4 v8f accumulators,
// A-fragment reused across 4 WMMAs per K-slab).
// Fast path: unchecked vectorized staging for full 32-wide K slabs;
// checked tail only for K % 32 != 0 (K=135, K=3).
// ---------------------------------------------------------------------------
#define LDSA_STRIDE 40   // halves; row = 80B (16B aligned)
#define LDSB_STRIDE 40

__global__ __launch_bounds__(256)
void gemm_wmma_kernel(const float* __restrict__ A, int lda, int K,
                      const float* __restrict__ W, const float* __restrict__ bias,
                      const float* __restrict__ res, float* __restrict__ C, int N,
                      int act, const float* __restrict__ alpha_p)
{
    __shared__ _Float16 ldsA[128 * LDSA_STRIDE];
    __shared__ _Float16 ldsB[64  * LDSB_STRIDE];

    const int tid  = threadIdx.x;
    const int lane = tid & 31;
    const int wave = tid >> 5;
    const int n0 = blockIdx.x * 64;
    const int m0 = blockIdx.y * 128;

    v8f acc0 = {}, acc1 = {}, acc2 = {}, acc3 = {};

    // cooperative staging coordinates
    const int arow_l = tid >> 1;          // 0..127 : A row, 16 floats each
    const int aseg   = (tid & 1) * 16;    // k offset 0/16
    const int wrow_l = tid >> 2;          // 0..63  : W row, 8 floats each
    const int wseg   = (tid & 3) * 8;     // k offset 0/8/16/24

    // fragment coordinates (CDNA5 16-bit A/B layouts)
    const int frow = (lane & 15);
    const int kh   = lane >> 4;

    const int  kfull = K & ~31;
    const bool vecok = ((lda & 3) == 0) && ((K & 3) == 0);

    for (int k0 = 0; k0 < kfull; k0 += 32) {
        // ---- stage A (128 x 32), unchecked ----
        {
            const float* ar = A + (size_t)(m0 + arow_l) * lda + k0 + aseg;
            _Float16* dst = &ldsA[arow_l * LDSA_STRIDE + aseg];
            if (k0 + 32 < kfull) __builtin_prefetch(ar + 32, 0, 1);
            float f[16];
            if (vecok) {
                const float4* a4 = (const float4*)ar;
                #pragma unroll
                for (int i = 0; i < 4; i++) {
                    float4 t = a4[i];
                    f[4*i] = t.x; f[4*i+1] = t.y; f[4*i+2] = t.z; f[4*i+3] = t.w;
                }
            } else {
                #pragma unroll
                for (int i = 0; i < 16; i++) f[i] = ar[i];
            }
            v8h h0, h1;
            #pragma unroll
            for (int i = 0; i < 8; i++) { h0[i] = (_Float16)f[i]; h1[i] = (_Float16)f[8+i]; }
            *(v8h*)&dst[0] = h0;
            *(v8h*)&dst[8] = h1;
        }
        // ---- stage W (64 x 32), unchecked K, guarded N ----
        {
            int n = n0 + wrow_l;
            float f[8];
            if (n < N) {
                const float* wr = W + (size_t)n * K + k0 + wseg;
                if (vecok) {
                    const float4* w4 = (const float4*)wr;
                    float4 t0 = w4[0], t1 = w4[1];
                    f[0]=t0.x; f[1]=t0.y; f[2]=t0.z; f[3]=t0.w;
                    f[4]=t1.x; f[5]=t1.y; f[6]=t1.z; f[7]=t1.w;
                } else {
                    #pragma unroll
                    for (int i = 0; i < 8; i++) f[i] = wr[i];
                }
            } else {
                #pragma unroll
                for (int i = 0; i < 8; i++) f[i] = 0.f;
            }
            v8h h;
            #pragma unroll
            for (int i = 0; i < 8; i++) h[i] = (_Float16)f[i];
            *(v8h*)&ldsB[wrow_l * LDSB_STRIDE + wseg] = h;
        }
        __syncthreads();

        // ---- fragments + 4 WMMAs ----
        {
            const int arow = wave * 16 + frow;
            v8h alo = *(const v8h*)&ldsA[arow * LDSA_STRIDE + kh * 16];
            v8h ahi = *(const v8h*)&ldsA[arow * LDSA_STRIDE + kh * 16 + 8];
            v16h af = __builtin_shufflevector(alo, ahi,
                         0,1,2,3,4,5,6,7,8,9,10,11,12,13,14,15);
            v16h b0 = *(const v16h*)&ldsB[( 0 + frow) * LDSB_STRIDE + kh * 16];
            v16h b1 = *(const v16h*)&ldsB[(16 + frow) * LDSB_STRIDE + kh * 16];
            v16h b2 = *(const v16h*)&ldsB[(32 + frow) * LDSB_STRIDE + kh * 16];
            v16h b3 = *(const v16h*)&ldsB[(48 + frow) * LDSB_STRIDE + kh * 16];
            acc0 = __builtin_amdgcn_wmma_f32_16x16x32_f16(false, af, false, b0, (short)0, acc0, false, false);
            acc1 = __builtin_amdgcn_wmma_f32_16x16x32_f16(false, af, false, b1, (short)0, acc1, false, false);
            acc2 = __builtin_amdgcn_wmma_f32_16x16x32_f16(false, af, false, b2, (short)0, acc2, false, false);
            acc3 = __builtin_amdgcn_wmma_f32_16x16x32_f16(false, af, false, b3, (short)0, acc3, false, false);
        }
        __syncthreads();
    }

    // ---- checked tail slab (only for K % 32 != 0) ----
    if (kfull < K) {
        {
            const float* ar = A + (size_t)(m0 + arow_l) * lda + kfull + aseg;
            _Float16* dst = &ldsA[arow_l * LDSA_STRIDE + aseg];
            #pragma unroll
            for (int i = 0; i < 16; i++) {
                int k = kfull + aseg + i;
                dst[i] = (_Float16)(k < K ? ar[i] : 0.f);
            }
        }
        {
            int n = n0 + wrow_l;
            _Float16* dst = &ldsB[wrow_l * LDSB_STRIDE + wseg];
            #pragma unroll
            for (int i = 0; i < 8; i++) {
                int k = kfull + wseg + i;
                float v = 0.f;
                if (n < N && k < K) v = W[(size_t)n * K + k];
                dst[i] = (_Float16)v;
            }
        }
        __syncthreads();
        {
            const int arow = wave * 16 + frow;
            v8h alo = *(const v8h*)&ldsA[arow * LDSA_STRIDE + kh * 16];
            v8h ahi = *(const v8h*)&ldsA[arow * LDSA_STRIDE + kh * 16 + 8];
            v16h af = __builtin_shufflevector(alo, ahi,
                         0,1,2,3,4,5,6,7,8,9,10,11,12,13,14,15);
            v16h b0 = *(const v16h*)&ldsB[( 0 + frow) * LDSB_STRIDE + kh * 16];
            v16h b1 = *(const v16h*)&ldsB[(16 + frow) * LDSB_STRIDE + kh * 16];
            v16h b2 = *(const v16h*)&ldsB[(32 + frow) * LDSB_STRIDE + kh * 16];
            v16h b3 = *(const v16h*)&ldsB[(48 + frow) * LDSB_STRIDE + kh * 16];
            acc0 = __builtin_amdgcn_wmma_f32_16x16x32_f16(false, af, false, b0, (short)0, acc0, false, false);
            acc1 = __builtin_amdgcn_wmma_f32_16x16x32_f16(false, af, false, b1, (short)0, acc1, false, false);
            acc2 = __builtin_amdgcn_wmma_f32_16x16x32_f16(false, af, false, b2, (short)0, acc2, false, false);
            acc3 = __builtin_amdgcn_wmma_f32_16x16x32_f16(false, af, false, b3, (short)0, acc3, false, false);
        }
        __syncthreads();
    }

    // ---- epilogue: bias + residual + activation ----
    const float alpha = alpha_p ? *alpha_p : 0.f;
    v8f accs[4] = { acc0, acc1, acc2, acc3 };
    #pragma unroll
    for (int nn = 0; nn < 4; nn++) {
        int nc = n0 + nn * 16 + frow;
        if (nc >= N) continue;
        float bv = bias ? bias[nc] : 0.f;
        #pragma unroll
        for (int r = 0; r < 8; r++) {
            int mr = m0 + wave * 16 + kh * 8 + r;   // D-matrix row map
            float v = accs[nn][r] + bv;
            if (res) v += res[(size_t)mr * N + nc];
            if      (act == ACT_PRELU) v = (v >= 0.f) ? v : alpha * v;
            else if (act == ACT_RELU)  v = fmaxf(v, 0.f);
            else if (act == ACT_SILU)  v = v / (1.f + __expf(-v));
            C[(size_t)mr * N + nc] = v;
        }
    }
}

// ---------------------------------------------------------------------------
// rel_enc: compact 31x64 table
// ---------------------------------------------------------------------------
__global__ void relenc_kernel(const float* __restrict__ w1, const float* __restrict__ b1,
                              const float* __restrict__ a1p,
                              const float* __restrict__ w2, const float* __restrict__ b2,
                              float* __restrict__ relc)
{
    int idx = blockIdx.x * blockDim.x + threadIdx.x;
    if (idx >= 31 * D_HEAD) return;
    int r = idx / D_HEAD, o = idx % D_HEAD;
    float rin = (float)(r - WIN_HALF);
    float a1  = *a1p;
    float acc = b2[o];
    for (int h = 0; h < D_MODEL; h++) {
        float t = w1[h] * rin + b1[h];
        t = (t >= 0.f) ? t : a1 * t;
        acc += t * w2[(size_t)o * D_MODEL + h];
    }
    relc[idx] = acc;
}

// ---------------------------------------------------------------------------
// Windowed MHA: one thread per (b,h,i); 31-key window; bias folded into dot.
// ---------------------------------------------------------------------------
__global__ __launch_bounds__(256)
void attn_kernel(const float* __restrict__ Q, const float* __restrict__ Kt,
                 const float* __restrict__ V, const float* __restrict__ relc,
                 float* __restrict__ O)
{
    int gid = blockIdx.x * blockDim.x + threadIdx.x;
    if (gid >= BB * N_HEADS * TT) return;
    int i = gid % TT;
    int h = (gid / TT) % N_HEADS;
    int b = gid / (TT * N_HEADS);

    const float* qp = Q + ((size_t)(b * TT + i)) * D_MODEL + h * D_HEAD;
    float q[D_HEAD];
    #pragma unroll
    for (int d = 0; d < D_HEAD; d++) q[d] = qp[d];

    float sc[2 * WIN_HALF + 1];
    float mx = -1e30f;
    #pragma unroll 4
    for (int dj = -WIN_HALF; dj <= WIN_HALF; dj++) {
        int j = i + dj;
        float s = -1e30f;
        if (j >= 0 && j < TT) {
            const float* kp = Kt + ((size_t)(b * TT + j)) * D_MODEL + h * D_HEAD;
            const float* rp = relc + (size_t)(dj + WIN_HALF) * D_HEAD;
            float acc = 0.f;
            #pragma unroll
            for (int d = 0; d < D_HEAD; d++) acc += q[d] * (kp[d] + rp[d]);
            s = acc * 0.125f;   // 1/sqrt(64)
        }
        sc[dj + WIN_HALF] = s;
        mx = fmaxf(mx, s);
    }
    float sum = 0.f;
    #pragma unroll
    for (int t = 0; t < 2 * WIN_HALF + 1; t++) {
        float e = __expf(sc[t] - mx); sc[t] = e; sum += e;
    }
    float inv = 1.f / sum;

    float o[D_HEAD];
    #pragma unroll
    for (int d = 0; d < D_HEAD; d++) o[d] = 0.f;
    for (int dj = -WIN_HALF; dj <= WIN_HALF; dj++) {
        int j = i + dj;
        if (j < 0 || j >= TT) continue;
        float w = sc[dj + WIN_HALF] * inv;
        const float* vp = V + ((size_t)(b * TT + j)) * D_MODEL + h * D_HEAD;
        #pragma unroll
        for (int d = 0; d < D_HEAD; d++) o[d] += w * vp[d];
    }
    float* op = O + ((size_t)(b * TT + i)) * D_MODEL + h * D_HEAD;
    #pragma unroll
    for (int d = 0; d < D_HEAD; d++) op[d] = o[d];
}

// ---------------------------------------------------------------------------
// Final split: out = (m[4096,135], sigmoid(s)[4096,1]) flat-concat
// ---------------------------------------------------------------------------
__global__ void split_kernel(const float* __restrict__ dec, float* __restrict__ out) {
    int idx = blockIdx.x * blockDim.x + threadIdx.x;
    const int total = MTOK * 136;
    if (idx >= total) return;
    int m = idx / 136, c = idx % 136;
    float v = dec[idx];
    if (c < D_MOTION) out[(size_t)m * D_MOTION + c] = v;
    else              out[(size_t)MTOK * D_MOTION + m] = 1.f / (1.f + __expf(-v));
}

// ---------------------------------------------------------------------------
// Host launch. Param leaves follow jax tree order (dict keys sorted).
// ---------------------------------------------------------------------------
extern "C" void kernel_launch(void* const* d_in, const int* in_sizes, int n_in,
                              void* d_out, int out_size, void* d_ws, size_t ws_size,
                              hipStream_t stream) {
    const float* x     = (const float*)d_in[0];
    const float* noise = (const float*)d_in[1];
    auto P = [&](int i) { return (const float*)d_in[i]; };

    float* ws = (float*)d_ws;
    size_t off = 0;
    auto alloc = [&](size_t n) { float* p = ws + off; off += n; return p; };
    float* motion    = alloc((size_t)MTOK * D_MODEL);
    float* traj      = alloc((size_t)MTOK * D_MODEL);
    float* xn        = alloc((size_t)MTOK * D_MODEL);
    float* cn        = alloc((size_t)MTOK * D_MODEL);
    float* qb        = alloc((size_t)MTOK * D_MODEL);
    float* kb        = alloc((size_t)MTOK * D_MODEL);
    float* vb        = alloc((size_t)MTOK * D_MODEL);
    float* attno     = alloc((size_t)MTOK * D_MODEL);
    float* ffh       = alloc((size_t)MTOK * D_FF);     // also enc/dec scratch
    float* motion_in = alloc((size_t)MTOK * D_MOTION);
    float* traj_in   = alloc((size_t)MTOK * 3);
    float* relc      = alloc((size_t)31 * D_HEAD);

    auto gemm = [&](const float* A, int lda, int K, const float* W, const float* bias,
                    const float* res, float* C, int N, int act, const float* alpha) {
        dim3 g((N + 63) / 64, MTOK / 128);
        hipLaunchKernelGGL(gemm_wmma_kernel, g, dim3(256), 0, stream,
                           A, lda, K, W, bias, res, C, N, act, alpha);
    };
    auto ln = [&](const float* X, const float* g, const float* b, float* Y) {
        hipLaunchKernelGGL(ln_kernel, dim3(MTOK / 8), dim3(256), 0, stream, X, g, b, Y);
    };
    auto attn = [&](const float* Q, const float* K, const float* V, float* O) {
        hipLaunchKernelGGL(attn_kernel, dim3((BB * N_HEADS * TT + 255) / 256),
                           dim3(256), 0, stream, Q, K, V, relc, O);
    };

    // --- prep + rel table ---
    hipLaunchKernelGGL(prep_kernel, dim3((MTOK * 138 + 255) / 256), dim3(256), 0, stream,
                       x, noise, motion_in, traj_in);
    hipLaunchKernelGGL(relenc_kernel, dim3((31 * D_HEAD + 255) / 256), dim3(256), 0, stream,
                       P(171), P(172), P(170), P(173), P(174), relc);

    // --- encoders ---
    gemm(motion_in, D_MOTION, D_MOTION, P(166), P(167), nullptr, ffh, D_MODEL, ACT_PRELU, P(164));
    gemm(ffh, D_MODEL, D_MODEL,         P(168), P(169), nullptr, motion, D_MODEL, ACT_PRELU, P(165));
    gemm(traj_in, 3, 3,                 P(177), P(178), nullptr, ffh, D_MODEL, ACT_PRELU, P(175));
    gemm(ffh, D_MODEL, D_MODEL,         P(179), P(180), nullptr, traj, D_MODEL, ACT_PRELU, P(176));

    // --- transformer layers ---
    for (int L = 0; L < N_LAYERS; L++) {
        int B0 = 6 + 26 * L;
        // self-attention (leaves +16..+25)
        {
            const float *Wk = P(B0+16), *Wkb = P(B0+17), *Wo = P(B0+18), *Wob = P(B0+19),
                        *Wq = P(B0+20), *Wqb = P(B0+21), *Wv = P(B0+22), *Wvb = P(B0+23),
                        *lnb = P(B0+24), *lng = P(B0+25);
            ln(motion, lng, lnb, xn);
            gemm(xn, D_MODEL, D_MODEL, Wq, Wqb, nullptr, qb, D_MODEL, ACT_NONE, nullptr);
            gemm(xn, D_MODEL, D_MODEL, Wk, Wkb, nullptr, kb, D_MODEL, ACT_NONE, nullptr);
            gemm(xn, D_MODEL, D_MODEL, Wv, Wvb, nullptr, vb, D_MODEL, ACT_NONE, nullptr);
            attn(qb, kb, vb, attno);
            gemm(attno, D_MODEL, D_MODEL, Wo, Wob, motion, motion, D_MODEL, ACT_NONE, nullptr);
        }
        // cross-attention (leaves +0..+9), context = traj
        {
            const float *Wk = P(B0+0), *Wkb = P(B0+1), *Wo = P(B0+2), *Wob = P(B0+3),
                        *Wq = P(B0+4), *Wqb = P(B0+5), *Wv = P(B0+6), *Wvb = P(B0+7),
                        *lnb = P(B0+8), *lng = P(B0+9);
            ln(motion, lng, lnb, xn);
            ln(traj,   lng, lnb, cn);
            gemm(xn, D_MODEL, D_MODEL, Wq, Wqb, nullptr, qb, D_MODEL, ACT_NONE, nullptr);
            gemm(cn, D_MODEL, D_MODEL, Wk, Wkb, nullptr, kb, D_MODEL, ACT_NONE, nullptr);
            gemm(cn, D_MODEL, D_MODEL, Wv, Wvb, nullptr, vb, D_MODEL, ACT_NONE, nullptr);
            attn(qb, kb, vb, attno);
            gemm(attno, D_MODEL, D_MODEL, Wo, Wob, motion, motion, D_MODEL, ACT_NONE, nullptr);
        }
        // pffn (leaves +10..+15)
        {
            const float *l1W = P(B0+10), *l1b = P(B0+11), *l2W = P(B0+12), *l2b = P(B0+13),
                        *lnb = P(B0+14), *lng = P(B0+15);
            ln(motion, lng, lnb, xn);
            gemm(xn, D_MODEL, D_MODEL, l1W, l1b, nullptr, ffh, D_FF, ACT_RELU, nullptr);
            gemm(ffh, D_FF, D_FF,      l2W, l2b, motion, motion, D_MODEL, ACT_NONE, nullptr);
        }
    }

    // --- decoder ---
    ln(motion, P(163), P(162), xn);
    gemm(xn,  D_MODEL, D_MODEL, P(2), P(3), nullptr, ffh, D_MODEL, ACT_SILU, nullptr);
    gemm(ffh, D_MODEL, D_MODEL, P(4), P(5), nullptr, qb, D_MOTION + 1, ACT_NONE, nullptr);
    hipLaunchKernelGGL(split_kernel, dim3((MTOK * 136 + 255) / 256), dim3(256), 0, stream,
                       qb, (float*)d_out);
}
